// AttentiveLSTM_32229434589398
// MI455X (gfx1250) — matrix operand order repaired and checked by reading, not compiled, not run on hardware
//
#include <hip/hip_runtime.h>

typedef __bf16 bf16;
typedef __attribute__((ext_vector_type(16))) __bf16 v16bf;
typedef __attribute__((ext_vector_type(8)))  __bf16 v8bf;
typedef __attribute__((ext_vector_type(8)))  float  v8f;

constexpr int kH  = 768;    // hidden
constexpr int kNH = 12;     // heads
constexpr int kHD = 64;     // head dim
constexpr int kB  = 100;    // batch
constexpr int kT  = 256;    // seq len
constexpr float kEPS = 1e-5f;

// ---------------------------------------------------------------- reductions
__device__ inline float block_sum(float v, volatile float* red) {
    int tid = threadIdx.x;
#pragma unroll
    for (int o = 16; o; o >>= 1) v += __shfl_xor(v, o, 32);
    if ((tid & 31) == 0) red[tid >> 5] = v;
    __syncthreads();
    if (tid == 0) {
        float s = 0.f;
#pragma unroll
        for (int i = 0; i < 8; ++i) s += red[i];
        red[0] = s;
    }
    __syncthreads();
    float s = red[0];
    __syncthreads();
    return s;
}

// ------------------------------------------------ CDNA5 async LDS staging
// GLOBAL_LOAD_ASYNC_TO_LDS_B128: memory -> LDS without VGPR round-trip,
// tracked by ASYNCcnt (CDNA5 ISA).
__device__ inline void async_b128_to_lds(unsigned lds_addr, unsigned long long gaddr) {
    asm volatile("global_load_async_to_lds_b128 %0, %1, off"
                 :: "v"(lds_addr), "v"(gaddr)
                 : "memory");
}
__device__ inline void wait_async0() {
    asm volatile("s_wait_asynccnt 0x0" ::: "memory");
}

// Stage a 16-row x kH A-tile (bf16) into LDS via the async path.
// Rows past M are clamped to M-1: they only feed output rows that are
// never stored, so duplication is harmless and keeps addresses in-bounds.
__device__ inline void stage_a_tile_async(bf16* Atile, const bf16* A, int m0, int M) {
    unsigned ldsBase = (unsigned)(uintptr_t)Atile;
    constexpr int CPR = kH / 8;                  // b128 chunks per row (96)
    for (int i = threadIdx.x; i < 16 * CPR; i += 256) {
        int r = i / CPR, cc = i - r * CPR;
        int gm = m0 + r;
        if (gm >= M) gm = M - 1;
        unsigned long long g =
            (unsigned long long)(uintptr_t)(A + (size_t)gm * kH + (size_t)cc * 8);
        async_b128_to_lds(ldsBase + (unsigned)i * 16u, g);
    }
    wait_async0();
    __syncthreads();
}

// --------------------------------------------------- WMMA fragment loaders
__device__ inline v16bf concat8(v8bf lo, v8bf hi) {
    return __builtin_shufflevector(lo, hi, 0, 1, 2, 3, 4, 5, 6, 7,
                                           8, 9, 10, 11, 12, 13, 14, 15);
}

// A: 16x32 bf16 (MxK) tile, row-major, leading dim lda. Per the ISA VGPR
// layout a lane's fragment is two contiguous 8-element runs -> 2x ds_load_b128.
__device__ inline v16bf load_a_frag(const bf16* A, int lda, int lane) {
    int hlf = (lane >> 4) & 1;
    int row = lane & 15;
    const bf16* p = A + row * lda + hlf * 8;
    v8bf lo = *(const v8bf*)(p);
    v8bf hi = *(const v8bf*)(p + 16);
    return concat8(lo, hi);
}

// B fragment from a [N][K] (row = output column) bf16 matrix:
// two 8-element runs per lane (k = 8h.., 16+8h..) -> 2x global_load_b128.
__device__ inline v16bf load_w_frag(const bf16* Wblk, int lane) {
    int colx = lane & 15;
    int kb   = (lane >> 4) << 3;
    const bf16* p = Wblk + colx * kH + kb;
    v8bf lo = *(const v8bf*)(p);
    v8bf hi = *(const v8bf*)(p + 16);
    return concat8(lo, hi);
}

// ------------------------------------------------------------ prep kernels
__global__ void k_convert(const float* __restrict__ in, bf16* __restrict__ out, int n) {
    int i = blockIdx.x * blockDim.x + threadIdx.x;
    if (i < n) out[i] = (bf16)in[i];
}

// in: [K][N] row-major fp32 ; out: [N][K] bf16 (transposed for b128 B-frag loads)
__global__ void k_convert_t(const float* __restrict__ in, bf16* __restrict__ out,
                            int K, int N) {
    int i = blockIdx.x * blockDim.x + threadIdx.x;
    if (i < K * N) {
        int k = i / N, n = i - k * N;
        out[n * K + k] = (bf16)in[i];
    }
}

__global__ void k_zero(float* __restrict__ p, int n) {
    int i = blockIdx.x * blockDim.x + threadIdx.x;
    if (i < n) p[i] = 0.f;
}

// ------------------------------------------------- per-step LN input kernel
// blocks 0..99  : ARow[b]      = LN1(LNe(emb[ids[b,t]]))
// blocks 100..199: ARow[100+b] = LN1(h[b])
__global__ void k_ln_inputs(int t, const int* __restrict__ ids,
                            const float* __restrict__ emb,
                            const float* __restrict__ lne_g, const float* __restrict__ lne_b,
                            const float* __restrict__ ln1_g, const float* __restrict__ ln1_b,
                            const float* __restrict__ h, bf16* __restrict__ ARow) {
    __shared__ float red[8];
    int tid = threadIdx.x;
    int blk = blockIdx.x;
    int j0 = tid, j1 = tid + 256, j2 = tid + 512;
    float x0, x1, x2;
    if (blk < kB) {
        int row = ids[blk * kT + t];
        const float* s = emb + row * kH;
        x0 = s[j0]; x1 = s[j1]; x2 = s[j2];
        float m  = block_sum(x0 + x1 + x2, red) * (1.f / kH);
        float d0 = x0 - m, d1 = x1 - m, d2 = x2 - m;
        float v  = block_sum(d0 * d0 + d1 * d1 + d2 * d2, red) * (1.f / kH);
        float rs = rsqrtf(v + kEPS);
        x0 = d0 * rs * lne_g[j0] + lne_b[j0];
        x1 = d1 * rs * lne_g[j1] + lne_b[j1];
        x2 = d2 * rs * lne_g[j2] + lne_b[j2];
    } else {
        const float* s = h + (blk - kB) * kH;
        x0 = s[j0]; x1 = s[j1]; x2 = s[j2];
    }
    float m  = block_sum(x0 + x1 + x2, red) * (1.f / kH);
    float d0 = x0 - m, d1 = x1 - m, d2 = x2 - m;
    float v  = block_sum(d0 * d0 + d1 * d1 + d2 * d2, red) * (1.f / kH);
    float rs = rsqrtf(v + kEPS);
    bf16* o = ARow + blk * kH;
    o[j0] = (bf16)(d0 * rs * ln1_g[j0] + ln1_b[j0]);
    o[j1] = (bf16)(d1 * rs * ln1_g[j1] + ln1_b[j1]);
    o[j2] = (bf16)(d2 * rs * ln1_g[j2] + ln1_b[j2]);
}

__global__ void k_ln_row(const float* __restrict__ X, const float* __restrict__ g,
                         const float* __restrict__ b, bf16* __restrict__ O) {
    __shared__ float red[8];
    int tid = threadIdx.x;
    int row = blockIdx.x;
    const float* s = X + row * kH;
    float x0 = s[tid], x1 = s[tid + 256], x2 = s[tid + 512];
    float m  = block_sum(x0 + x1 + x2, red) * (1.f / kH);
    float d0 = x0 - m, d1 = x1 - m, d2 = x2 - m;
    float v  = block_sum(d0 * d0 + d1 * d1 + d2 * d2, red) * (1.f / kH);
    float rs = rsqrtf(v + kEPS);
    bf16* o = O + row * kH;
    o[tid]       = (bf16)(d0 * rs * g[tid] + b[tid]);
    o[tid + 256] = (bf16)(d1 * rs * g[tid + 256] + b[tid + 256]);
    o[tid + 512] = (bf16)(d2 * rs * g[tid + 512] + b[tid + 512]);
}

// ----------------------------------------------------------- WMMA GEMM
// Out[M,N] = A[M,768](bf16) * Wt[N,768](bf16, pre-transposed) [+bias] [+resid]
// grid: (N/128, ceil(M/16)), block 256 (8 waves, one 16x16 tile each)
template <bool HAS_BIAS, bool HAS_RES>
__global__ void __launch_bounds__(256) k_gemm(const bf16* __restrict__ A, int M,
                                              const bf16* __restrict__ Wt, int N,
                                              const float* __restrict__ bias,
                                              const float* __restrict__ resid,
                                              float* __restrict__ Out) {
    __shared__ alignas(16) bf16 Atile[16 * kH];
    int m0 = blockIdx.y * 16;
    stage_a_tile_async(Atile, A, m0, M);

    int lane = threadIdx.x & 31;
    int wave = threadIdx.x >> 5;
    int n0   = blockIdx.x * 128 + wave * 16;
    const bf16* Wbase = Wt + (size_t)n0 * kH;
    v8f acc = {};
#pragma unroll 4
    for (int k = 0; k < kH; k += 32) {
        v16bf a = load_a_frag(&Atile[k], kH, lane);
        v16bf b = load_w_frag(Wbase + k, lane);
        acc = __builtin_amdgcn_wmma_f32_16x16x32_bf16(false, a, false, b,
                                                      (short)0, acc, false, false);
    }
    int col = n0 + (lane & 15);
    int mb  = (lane >> 4) * 8;
    float bcol = HAS_BIAS ? bias[col] : 0.f;
#pragma unroll
    for (int r = 0; r < 8; ++r) {
        int m = m0 + mb + r;
        if (m < M) {
            float v = acc[r] + bcol;
            if (HAS_RES) v += resid[m * N + col];
            Out[m * N + col] = v;
        }
    }
}

// ------------------------------------------------------- 2-key attention
__global__ void k_attn(const float* __restrict__ q, const float* __restrict__ kv,
                       bf16* __restrict__ ctx) {
    int gtid = blockIdx.x * blockDim.x + threadIdx.x;
    int w    = gtid >> 5;
    int lane = gtid & 31;
    if (w >= kB * kNH) return;
    int b = w / kNH, hseg = w % kNH;
    int d = lane * 2;
    int qi = b * kH + hseg * kHD + d;
    float q0 = q[qi], q1 = q[qi + 1];
    const float* kve = kv + b * (2 * kH);
    const float* kvh = kv + (kB + b) * (2 * kH);
    int ki = hseg * kHD + d;
    float s0 = q0 * kve[ki] + q1 * kve[ki + 1];
    float s1 = q0 * kvh[ki] + q1 * kvh[ki + 1];
#pragma unroll
    for (int o = 16; o; o >>= 1) {
        s0 += __shfl_xor(s0, o, 32);
        s1 += __shfl_xor(s1, o, 32);
    }
    s0 *= 0.125f; s1 *= 0.125f;                  // 1/sqrt(64)
    float mx = fmaxf(s0, s1);
    float p0 = __expf(s0 - mx), p1 = __expf(s1 - mx);
    float inv = 1.f / (p0 + p1);
    p0 *= inv; p1 *= inv;
    int vi = kH + hseg * kHD + d;
    ctx[qi]     = (bf16)(p0 * kve[vi]     + p1 * kvh[vi]);
    ctx[qi + 1] = (bf16)(p0 * kve[vi + 1] + p1 * kvh[vi + 1]);
}

// ---------------------------------------------- gates GEMM + LSTM cell fused
// gates = sigmoid(X @ Whh^T + bhh); Whh is [4H][768] = [N][K] already.
__global__ void __launch_bounds__(256) k_gates(const bf16* __restrict__ X,
                                               const bf16* __restrict__ Whh,
                                               const float* __restrict__ bhh,
                                               float* __restrict__ h, float* __restrict__ c,
                                               float* __restrict__ Out, int t) {
    __shared__ alignas(16) bf16 Atile[16 * kH];
    int m0 = blockIdx.y * 16;
    stage_a_tile_async(Atile, X, m0, kB);

    int lane = threadIdx.x & 31;
    int wave = threadIdx.x >> 5;
    int n0   = blockIdx.x * 128 + wave * 16;
    const bf16* WI = Whh + (size_t)(0 * kH + n0) * kH;
    const bf16* WF = Whh + (size_t)(1 * kH + n0) * kH;
    const bf16* WG = Whh + (size_t)(2 * kH + n0) * kH;
    const bf16* WO = Whh + (size_t)(3 * kH + n0) * kH;
    v8f accI = {}, accF = {}, accG = {}, accO = {};
#pragma unroll 2
    for (int k = 0; k < kH; k += 32) {
        v16bf a   = load_a_frag(&Atile[k], kH, lane);
        v16bf bi  = load_w_frag(WI + k, lane);
        v16bf bf_ = load_w_frag(WF + k, lane);
        v16bf bg  = load_w_frag(WG + k, lane);
        v16bf bo  = load_w_frag(WO + k, lane);
        accI = __builtin_amdgcn_wmma_f32_16x16x32_bf16(false, a, false, bi,  (short)0, accI, false, false);
        accF = __builtin_amdgcn_wmma_f32_16x16x32_bf16(false, a, false, bf_, (short)0, accF, false, false);
        accG = __builtin_amdgcn_wmma_f32_16x16x32_bf16(false, a, false, bg,  (short)0, accG, false, false);
        accO = __builtin_amdgcn_wmma_f32_16x16x32_bf16(false, a, false, bo,  (short)0, accO, false, false);
    }
    int col = n0 + (lane & 15);
    int mb  = (lane >> 4) * 8;
    float bI = bhh[col];
    float bF = bhh[col + kH];
    float bG = bhh[col + 2 * kH];
    float bO = bhh[col + 3 * kH];
#pragma unroll
    for (int r = 0; r < 8; ++r) {
        int m = m0 + mb + r;
        if (m < kB) {
            float gi = 1.f / (1.f + __expf(-(accI[r] + bI)));
            float gf = 1.f / (1.f + __expf(-(accF[r] + bF)));
            float gg = 1.f / (1.f + __expf(-(accG[r] + bG)));
            float go = 1.f / (1.f + __expf(-(accO[r] + bO)));
            float cold = c[m * kH + col];
            float cy = gf * cold + gi * gg;
            float hy = go * tanhf(cy);
            c[m * kH + col] = cy;
            h[m * kH + col] = hy;
            Out[(m * kT + t) * kH + col] = hy;
        }
    }
}

// ---------------------------------------------------------------- launcher
extern "C" void kernel_launch(void* const* d_in, const int* in_sizes, int n_in,
                              void* d_out, int out_size, void* d_ws, size_t ws_size,
                              hipStream_t stream) {
    (void)in_sizes; (void)n_in; (void)out_size; (void)ws_size;
    const int*   ids   = (const int*)d_in[0];
    const float* emb   = (const float*)d_in[1];
    const float* lne_g = (const float*)d_in[2];
    const float* lne_b = (const float*)d_in[3];
    const float* ln1_g = (const float*)d_in[4];
    const float* ln1_b = (const float*)d_in[5];
    const float* ln2_g = (const float*)d_in[6];
    const float* ln2_b = (const float*)d_in[7];
    const float* Wkv   = (const float*)d_in[8];
    const float* bkv   = (const float*)d_in[9];
    const float* Wq    = (const float*)d_in[10];
    const float* bq    = (const float*)d_in[11];
    const float* Wc    = (const float*)d_in[12];
    const float* bc    = (const float*)d_in[13];
    const float* Whh   = (const float*)d_in[14];
    const float* bhh   = (const float*)d_in[15];
    float* Out = (float*)d_out;

    char* p = (char*)d_ws;
    auto alloc = [&](size_t bytes) -> char* {
        char* r = p;
        p += (bytes + 255) & ~(size_t)255;
        return r;
    };
    bf16*  WqT    = (bf16*)alloc((size_t)kH * kH * 2);        // [N][K]
    bf16*  WcT    = (bf16*)alloc((size_t)kH * kH * 2);        // [N][K]
    bf16*  WkvT   = (bf16*)alloc((size_t)2 * kH * kH * 2);    // [2H][K]
    bf16*  Whh_bf = (bf16*)alloc((size_t)4 * kH * kH * 2);    // [4H][K] (native)
    bf16*  ARow   = (bf16*)alloc((size_t)2 * kB * kH * 2);    // [ebar_t ; hbar]
    float* qbuf   = (float*)alloc((size_t)kB * kH * 4);
    float* kvbuf  = (float*)alloc((size_t)2 * kB * 2 * kH * 4);
    bf16*  ctxb   = (bf16*)alloc((size_t)kB * kH * 2);
    float* ybuf   = (float*)alloc((size_t)kB * kH * 4);
    bf16*  xbar   = (bf16*)alloc((size_t)kB * kH * 2);
    float* hbuf   = (float*)alloc((size_t)kB * kH * 4);
    float* cbuf   = (float*)alloc((size_t)kB * kH * 4);

    // one-time prep: bf16 (transposed) weights + zero recurrent state
    int nWq = kH * kH, nWkv = kH * 2 * kH, nWhh = 4 * kH * kH;
    k_convert_t<<<(nWq  + 255) / 256, 256, 0, stream>>>(Wq,  WqT,  kH, kH);
    k_convert_t<<<(nWq  + 255) / 256, 256, 0, stream>>>(Wc,  WcT,  kH, kH);
    k_convert_t<<<(nWkv + 255) / 256, 256, 0, stream>>>(Wkv, WkvT, kH, 2 * kH);
    k_convert  <<<(nWhh + 255) / 256, 256, 0, stream>>>(Whh, Whh_bf, nWhh);
    int nstate = kB * kH;
    k_zero<<<(nstate + 255) / 256, 256, 0, stream>>>(hbuf, nstate);
    k_zero<<<(nstate + 255) / 256, 256, 0, stream>>>(cbuf, nstate);

    for (int t = 0; t < kT; ++t) {
        // LN1(LNe(emb[ids[:,t]])) and LN1(h) -> stacked bf16 A rows
        k_ln_inputs<<<2 * kB, 256, 0, stream>>>(t, ids, emb, lne_g, lne_b,
                                                ln1_g, ln1_b, hbuf, ARow);
        // q = hbar @ Wq + bq
        k_gemm<true, false><<<dim3(kH / 128, (kB + 15) / 16), 256, 0, stream>>>(
            ARow + kB * kH, kB, WqT, kH, bq, nullptr, qbuf);
        // kv = [ebar; hbar] @ Wkv + bkv   (200 x 1536)
        k_gemm<true, false><<<dim3(2 * kH / 128, (2 * kB + 15) / 16), 256, 0, stream>>>(
            ARow, 2 * kB, WkvT, 2 * kH, bkv, nullptr, kvbuf);
        // attention over the two keys -> ctx (bf16)
        k_attn<<<(kB * kNH * 32 + 127) / 128, 128, 0, stream>>>(qbuf, kvbuf, ctxb);
        // y = h + ctx @ Wc + bc
        k_gemm<true, true><<<dim3(kH / 128, (kB + 15) / 16), 256, 0, stream>>>(
            ctxb, kB, WcT, kH, bc, hbuf, ybuf);
        // x = LN2(y)
        k_ln_row<<<kB, 256, 0, stream>>>(ybuf, ln2_g, ln2_b, xbar);
        // gates + cell update + output write
        k_gates<<<dim3(kH / 128, (kB + 15) / 16), 256, 0, stream>>>(
            xbar, Whh_bf, bhh, hbuf, cbuf, Out, t);
    }
}
